// BitNetMoELayer_1340029796979
// MI455X (gfx1250) — 8-bit matrix operand order repaired and checked by reading, compiled, not run on hardware
//
#include <hip/hip_runtime.h>
#include <hip/hip_bf16.h>

typedef __attribute__((ext_vector_type(8))) int v8i;
typedef __attribute__((ext_vector_type(4))) int v4i;

#define T_TOK 8192
#define HDIM 768
#define IDIM 2048
#define NEXP 8
#define LDSTR 80

#if defined(__has_builtin)
#if __has_builtin(__builtin_amdgcn_global_load_async_to_lds_b128)
#define HAVE_ASYNC_LDS 1
#endif
#endif

#if defined(HAVE_ASYNC_LDS)
typedef __attribute__((address_space(1))) v4i gv4i;
typedef __attribute__((address_space(3))) v4i sv4i;
#endif

__device__ __forceinline__ float clipf(float v, float c){ return fminf(fmaxf(v,-c),c); }

__device__ __forceinline__ void copy16(unsigned char* l, const signed char* g){
#if defined(HAVE_ASYNC_LDS)
  __builtin_amdgcn_global_load_async_to_lds_b128((gv4i*)g, (sv4i*)l, 0, 0);
#else
  *(int4*)l = *(const int4*)g;
#endif
}
__device__ __forceinline__ void async_fence(){
#if defined(HAVE_ASYNC_LDS)
#if __has_builtin(__builtin_amdgcn_s_wait_asynccnt)
  __builtin_amdgcn_s_wait_asynccnt(0);
#else
  asm volatile("s_wait_asynccnt 0x0" ::: "memory");
#endif
#endif
}

__device__ __forceinline__ v8i load_a_frag(const unsigned char* l, int lane, int tileM){
  int hl = lane >> 4;
  int r  = tileM*16 + (lane & 15);
  v8i a;
#pragma unroll
  for (int j = 0; j < 8; ++j) {
    int koff = 16*(j>>1) + 4*(j&1) + 8*hl;
    a[j] = *(const int*)(l + r*LDSTR + koff);
  }
  return a;
}
__device__ __forceinline__ v8i load_b_frag(const unsigned char* l, int lane, int tileN){
  int hl = lane >> 4;
  int n  = tileN*16 + (lane & 15);
  v8i b;
#pragma unroll
  for (int j = 0; j < 8; ++j) {
    int koff = 16*(j>>1) + 4*(j&1) + 8*hl;
    b[j] = *(const int*)(l + n*LDSTR + koff);
  }
  return b;
}

__device__ __forceinline__ const float* pick_w(int mi,
    const float* Wg, const float* Wu, const float* Wd,
    const float* sWg, const float* sWu, const float* sWd){
  const size_t N = (size_t)IDIM * HDIM;
  if (mi < 24) { int e = mi/3, k = mi%3; return (k==0?Wg:(k==1?Wu:Wd)) + (size_t)e*N; }
  int k = mi - 24; return k==0?sWg:(k==1?sWu:sWd);
}

__global__ __launch_bounds__(256)
void wquant_partial_kernel(const float* __restrict__ Wg, const float* __restrict__ Wu,
                           const float* __restrict__ Wd, const float* __restrict__ sWg,
                           const float* __restrict__ sWu, const float* __restrict__ sWd,
                           float* __restrict__ wpart)
{
  int mi = blockIdx.y, blk = blockIdx.x, tid = threadIdx.x;
  const float* src = pick_w(mi, Wg, Wu, Wd, sWg, sWu, sWd);
  const float4* s4 = (const float4*)(src + (size_t)blk * 24576);
  float s = 0.f, sa = 0.f;
  for (int i = tid; i < 6144; i += 256) {
    float4 w = s4[i];
    s  += w.x + w.y + w.z + w.w;
    sa += fabsf(w.x) + fabsf(w.y) + fabsf(w.z) + fabsf(w.w);
  }
  __shared__ float r1[256], r2[256];
  r1[tid] = s; r2[tid] = sa; __syncthreads();
  for (int st = 128; st > 0; st >>= 1) {
    if (tid < st) { r1[tid] += r1[tid+st]; r2[tid] += r2[tid+st]; }
    __syncthreads();
  }
  if (tid == 0) { wpart[(mi*64+blk)*2] = r1[0]; wpart[(mi*64+blk)*2+1] = r2[0]; }
}

__global__ __launch_bounds__(256)
void wquant_final_kernel(const float* __restrict__ Wg, const float* __restrict__ Wu,
                         const float* __restrict__ Wd, const float* __restrict__ sWg,
                         const float* __restrict__ sWu, const float* __restrict__ sWd,
                         const float* __restrict__ wpart,
                         signed char* __restrict__ qw, float* __restrict__ wscale)
{
  int mi = blockIdx.y, blk = blockIdx.x, tid = threadIdx.x;
  const float* src = pick_w(mi, Wg, Wu, Wd, sWg, sWu, sWd);
  float s = 0.f, sa = 0.f;
  for (int i = 0; i < 64; ++i) { s += wpart[(mi*64+i)*2]; sa += wpart[(mi*64+i)*2+1]; }
  const size_t N = (size_t)IDIM * HDIM;
  float mean  = s / (float)N;
  float scale = fmaxf(sa / (float)N, 1e-8f);
  const float4* s4 = (const float4*)(src + (size_t)blk * 24576);
  char4* dst = (char4*)(qw + (size_t)mi * N + (size_t)blk * 24576);
  for (int i = tid; i < 6144; i += 256) {
    float4 w = s4[i];
    char4 q;
    float d;
    d = w.x - mean; q.x = (char)(d > 0.f ? 1 : (d < 0.f ? -1 : 0));
    d = w.y - mean; q.y = (char)(d > 0.f ? 1 : (d < 0.f ? -1 : 0));
    d = w.z - mean; q.z = (char)(d > 0.f ? 1 : (d < 0.f ? -1 : 0));
    d = w.w - mean; q.w = (char)(d > 0.f ? 1 : (d < 0.f ? -1 : 0));
    dst[i] = q;
  }
  if (blk == 0 && tid == 0) wscale[mi] = scale;
}

__global__ __launch_bounds__(256)
void router_kernel(const float* __restrict__ x, const float* __restrict__ rw,
                   const float* __restrict__ rbias, float* __restrict__ logits,
                   float* __restrict__ wtab)
{
  int tid = threadIdx.x, lane = tid & 31, wv = tid >> 5;
  int t = blockIdx.x * 8 + wv;
  float acc[NEXP];
#pragma unroll
  for (int e = 0; e < NEXP; ++e) acc[e] = 0.f;
  const float* xr = x + (size_t)t * HDIM;
  for (int k = lane; k < HDIM; k += 32) {
    float xv = xr[k];
#pragma unroll
    for (int e = 0; e < NEXP; ++e) acc[e] += xv * rw[e*HDIM + k];
  }
#pragma unroll
  for (int e = 0; e < NEXP; ++e)
    for (int o = 16; o > 0; o >>= 1) acc[e] += __shfl_xor(acc[e], o, 32);
  if (lane == 0) {
    float lg[NEXP], mx = -1e30f;
    for (int e = 0; e < NEXP; ++e) { lg[e] = acc[e] + rbias[e]; mx = fmaxf(mx, lg[e]); }
    float p[NEXP], den = 0.f;
    for (int e = 0; e < NEXP; ++e) { p[e] = __expf(lg[e]-mx); den += p[e]; }
    for (int e = 0; e < NEXP; ++e) logits[(size_t)t*NEXP + e] = lg[e];
    int i1 = 0;
    for (int e = 1; e < NEXP; ++e) if (p[e] > p[i1]) i1 = e;
    int i2 = (i1 == 0) ? 1 : 0;
    for (int e = 0; e < NEXP; ++e) { if (e == i1) continue; if (p[e] > p[i2]) i2 = e; }
    float s1 = p[i1]/den, s2 = p[i2]/den;
    float inv = 1.f / (s1 + s2 + 1e-8f);
    for (int e = 0; e < NEXP; ++e) {
      float w = (e == i1) ? s1*inv : (e == i2 ? s2*inv : 0.f);
      wtab[(size_t)e*T_TOK + t] = w;
    }
  }
}

__global__ __launch_bounds__(256)
void actquant_kernel(const float* __restrict__ x, int width, int nper,
                     const float* __restrict__ nw0, const float* __restrict__ nw1,
                     signed char* __restrict__ q0, signed char* __restrict__ q1,
                     float* __restrict__ s0, float* __restrict__ s1)
{
  __shared__ float red[256];
  int row = blockIdx.x, tid = threadIdx.x;
  const float* xr = x + (size_t)row * width;
  float v[8];
  float ss = 0.f;
  for (int i = 0; i < nper; ++i) {
    float xv = clipf(xr[tid + i*256], 100.f);
    v[i] = xv; ss += xv*xv;
  }
  red[tid] = ss; __syncthreads();
  for (int s = 128; s > 0; s >>= 1) { if (tid < s) red[tid] += red[tid+s]; __syncthreads(); }
  float var = fmaxf(red[0] / (float)width, 1e-5f);
  float inv = rsqrtf(var + 1e-5f);
  for (int i = 0; i < nper; ++i) v[i] = clipf(v[i]*inv, 10.f);

  const float*  nws[2] = { nw0, nw1 };
  signed char*  qs[2]  = { q0, q1 };
  float*        sc[2]  = { s0, s1 };
  for (int b = 0; b < 2; ++b) {
    if (!nws[b]) break;
    float a[8]; float mx = 0.f;
    for (int i = 0; i < nper; ++i) {
      int k = tid + i*256;
      a[i] = clipf(v[i]*nws[b][k], 50.f);
      mx = fmaxf(mx, fabsf(a[i]));
    }
    __syncthreads();
    red[tid] = mx; __syncthreads();
    for (int s = 128; s > 0; s >>= 1) { if (tid < s) red[tid] = fmaxf(red[tid], red[tid+s]); __syncthreads(); }
    float rowmax = fmaxf(red[0], 1e-4f);
    float scale  = 127.f / rowmax;
    for (int i = 0; i < nper; ++i) {
      int k = tid + i*256;
      float q = fminf(clipf(rintf(a[i]*scale), 128.f), 127.f);
      qs[b][(size_t)row*width + k] = (signed char)q;
    }
    if (tid == 0) sc[b][row] = rowmax / 127.f;
    __syncthreads();
  }
}

__global__ __launch_bounds__(256)
void gemm1_kernel(const signed char* __restrict__ xqg, const signed char* __restrict__ xqu,
                  const float* __restrict__ asg, const float* __restrict__ asu,
                  const signed char* __restrict__ qwg, const signed char* __restrict__ qwu,
                  const float* __restrict__ wscale, int mig, int miu,
                  float* __restrict__ hidden)
{
  __shared__ unsigned char lag[2][128*LDSTR], lau[2][128*LDSTR];
  __shared__ unsigned char lbg[2][ 64*LDSTR], lbu[2][ 64*LDSTR];
  int tid = threadIdx.x, lane = tid & 31, wv = tid >> 5;
  int wrow = wv >> 1, wcol = wv & 1;
  int m0 = blockIdx.x * 128, n0 = blockIdx.y * 64;
  v8i accg[2][2] = {}, accu[2][2] = {};

  auto load_tiles = [&](int buf, int kt){
#pragma unroll
    for (int ch = 0; ch < 2; ++ch) {
      int id = tid + ch*256;
      int r = id >> 2, c = id & 3;
      copy16(&lag[buf][r*LDSTR + c*16], xqg + (size_t)(m0+r)*HDIM + kt + c*16);
      copy16(&lau[buf][r*LDSTR + c*16], xqu + (size_t)(m0+r)*HDIM + kt + c*16);
    }
    int r = tid >> 2, c = tid & 3;
    copy16(&lbg[buf][r*LDSTR + c*16], qwg + (size_t)(n0+r)*HDIM + kt + c*16);
    copy16(&lbu[buf][r*LDSTR + c*16], qwu + (size_t)(n0+r)*HDIM + kt + c*16);
  };

  load_tiles(0, 0);
  async_fence();
  __syncthreads();
  for (int kt = 0, it = 0; kt < HDIM; kt += 64, ++it) {
    int cur = it & 1;
    if (kt + 64 < HDIM) load_tiles(cur ^ 1, kt + 64);
    {
      v8i a0 = load_a_frag(lag[cur], lane, 2*wrow+0);
      v8i a1 = load_a_frag(lag[cur], lane, 2*wrow+1);
      v8i b0 = load_b_frag(lbg[cur], lane, 2*wcol+0);
      v8i b1 = load_b_frag(lbg[cur], lane, 2*wcol+1);
      accg[0][0] = __builtin_amdgcn_wmma_i32_16x16x64_iu8(true, a0, true, b0, accg[0][0], false, false);
      accg[0][1] = __builtin_amdgcn_wmma_i32_16x16x64_iu8(true, a0, true, b1, accg[0][1], false, false);
      accg[1][0] = __builtin_amdgcn_wmma_i32_16x16x64_iu8(true, a1, true, b0, accg[1][0], false, false);
      accg[1][1] = __builtin_amdgcn_wmma_i32_16x16x64_iu8(true, a1, true, b1, accg[1][1], false, false);
    }
    {
      v8i a0 = load_a_frag(lau[cur], lane, 2*wrow+0);
      v8i a1 = load_a_frag(lau[cur], lane, 2*wrow+1);
      v8i b0 = load_b_frag(lbu[cur], lane, 2*wcol+0);
      v8i b1 = load_b_frag(lbu[cur], lane, 2*wcol+1);
      accu[0][0] = __builtin_amdgcn_wmma_i32_16x16x64_iu8(true, a0, true, b0, accu[0][0], false, false);
      accu[0][1] = __builtin_amdgcn_wmma_i32_16x16x64_iu8(true, a0, true, b1, accu[0][1], false, false);
      accu[1][0] = __builtin_amdgcn_wmma_i32_16x16x64_iu8(true, a1, true, b0, accu[1][0], false, false);
      accu[1][1] = __builtin_amdgcn_wmma_i32_16x16x64_iu8(true, a1, true, b1, accu[1][1], false, false);
    }
    async_fence();
    __syncthreads();
  }

  float wsg = wscale[mig], wsu = wscale[miu];
  int hl = lane >> 4;
#pragma unroll
  for (int mt = 0; mt < 2; ++mt)
#pragma unroll
  for (int nt = 0; nt < 2; ++nt) {
    int nn = n0 + (2*wcol+nt)*16 + (lane & 15);
#pragma unroll
    for (int j = 0; j < 8; ++j) {
      int m = m0 + (2*wrow+mt)*16 + hl*8 + j;
      float g = (float)accg[mt][nt][j] * asg[m] * wsg;
      g = clipf(g, 20.f);
      float u = (float)accu[mt][nt][j] * asu[m] * wsu;
      float h = (g / (1.f + __expf(-g))) * u;
      hidden[(size_t)m*IDIM + nn] = clipf(h, 1000.f);
    }
  }
}

__global__ __launch_bounds__(256)
void gemm2_kernel(const signed char* __restrict__ hq, const float* __restrict__ hs,
                  const signed char* __restrict__ qwd, const float* __restrict__ wscale,
                  int mid, const float* __restrict__ wrow, float* __restrict__ out)
{
  __shared__ unsigned char la[2][128*LDSTR], lb[2][64*LDSTR];
  int tid = threadIdx.x, lane = tid & 31, wv = tid >> 5;
  int wr = wv >> 1, wc = wv & 1;
  int m0 = blockIdx.x * 128, n0 = blockIdx.y * 64;
  v8i acc[2][2] = {};

  auto load_tiles = [&](int buf, int kt){
#pragma unroll
    for (int ch = 0; ch < 2; ++ch) {
      int id = tid + ch*256;
      int r = id >> 2, c = id & 3;
      copy16(&la[buf][r*LDSTR + c*16], hq + (size_t)(m0+r)*IDIM + kt + c*16);
    }
    int r = tid >> 2, c = tid & 3;
    copy16(&lb[buf][r*LDSTR + c*16], qwd + (size_t)(n0+r)*IDIM + kt + c*16);
  };

  load_tiles(0, 0);
  async_fence();
  __syncthreads();
  for (int kt = 0, it = 0; kt < IDIM; kt += 64, ++it) {
    int cur = it & 1;
    if (kt + 64 < IDIM) load_tiles(cur ^ 1, kt + 64);
    v8i a0 = load_a_frag(la[cur], lane, 2*wr+0);
    v8i a1 = load_a_frag(la[cur], lane, 2*wr+1);
    v8i b0 = load_b_frag(lb[cur], lane, 2*wc+0);
    v8i b1 = load_b_frag(lb[cur], lane, 2*wc+1);
    acc[0][0] = __builtin_amdgcn_wmma_i32_16x16x64_iu8(true, a0, true, b0, acc[0][0], false, false);
    acc[0][1] = __builtin_amdgcn_wmma_i32_16x16x64_iu8(true, a0, true, b1, acc[0][1], false, false);
    acc[1][0] = __builtin_amdgcn_wmma_i32_16x16x64_iu8(true, a1, true, b0, acc[1][0], false, false);
    acc[1][1] = __builtin_amdgcn_wmma_i32_16x16x64_iu8(true, a1, true, b1, acc[1][1], false, false);
    async_fence();
    __syncthreads();
  }

  float wsd = wscale[mid];
  int hl = lane >> 4;
#pragma unroll
  for (int mt = 0; mt < 2; ++mt)
#pragma unroll
  for (int nt = 0; nt < 2; ++nt) {
    int nn = n0 + (2*wc+nt)*16 + (lane & 15);
#pragma unroll
    for (int j = 0; j < 8; ++j) {
      int m = m0 + (2*wr+mt)*16 + hl*8 + j;
      float we = wrow ? wrow[m] : 1.f;
      float y  = (float)acc[mt][nt][j] * hs[m] * wsd * we;
      out[(size_t)m*HDIM + nn] += y;
    }
  }
}

__global__ void clip_kernel(float* __restrict__ o, int n) {
  int i = blockIdx.x * blockDim.x + threadIdx.x;
  if (i < n) o[i] = clipf(o[i], 10000.f);
}

extern "C" void kernel_launch(void* const* d_in, const int* in_sizes, int n_in,
                              void* d_out, int out_size, void* d_ws, size_t ws_size,
                              hipStream_t stream)
{
  (void)in_sizes; (void)n_in; (void)out_size; (void)ws_size;
  const float* x    = (const float*)d_in[0];
  const float* rw   = (const float*)d_in[1];
  const float* rb   = (const float*)d_in[2];
  const float* Wg   = (const float*)d_in[3];
  const float* Wu   = (const float*)d_in[4];
  const float* Wd   = (const float*)d_in[5];
  const float* ng   = (const float*)d_in[6];
  const float* nu   = (const float*)d_in[7];
  const float* nd   = (const float*)d_in[8];
  const float* sWg  = (const float*)d_in[9];
  const float* sWu  = (const float*)d_in[10];
  const float* sWd  = (const float*)d_in[11];
  const float* sng  = (const float*)d_in[12];
  const float* snu  = (const float*)d_in[13];
  const float* snd  = (const float*)d_in[14];

  float* out    = (float*)d_out;
  float* logits = out + (size_t)T_TOK * HDIM;

  unsigned char* p = (unsigned char*)d_ws;
  auto alloc = [&](size_t n) { unsigned char* q = p; p += (n + 255) & ~(size_t)255; return q; };
  const size_t WN = (size_t)IDIM * HDIM;
  float*       wscale = (float*)      alloc(27 * sizeof(float));
  float*       wpart  = (float*)      alloc(27 * 64 * 2 * sizeof(float));
  signed char* qw     = (signed char*)alloc(27 * WN);
  float*       wtab   = (float*)      alloc((size_t)NEXP * T_TOK * sizeof(float));
  signed char* xqg    = (signed char*)alloc((size_t)T_TOK * HDIM);
  signed char* xqu    = (signed char*)alloc((size_t)T_TOK * HDIM);
  float*       asg    = (float*)      alloc((size_t)T_TOK * sizeof(float));
  float*       asu    = (float*)      alloc((size_t)T_TOK * sizeof(float));
  float*       hidden = (float*)      alloc((size_t)T_TOK * IDIM * sizeof(float));
  signed char* hq     = (signed char*)alloc((size_t)T_TOK * IDIM);
  float*       hs     = (float*)      alloc((size_t)T_TOK * sizeof(float));

  (void)hipMemsetAsync(out, 0, (size_t)T_TOK * HDIM * sizeof(float), stream);
  wquant_partial_kernel<<<dim3(64,27), 256, 0, stream>>>(Wg, Wu, Wd, sWg, sWu, sWd, wpart);
  wquant_final_kernel  <<<dim3(64,27), 256, 0, stream>>>(Wg, Wu, Wd, sWg, sWu, sWd, wpart, qw, wscale);
  router_kernel<<<T_TOK/8, 256, 0, stream>>>(x, rw, rb, logits, wtab);

  for (int e = 0; e < 9; ++e) {
    const float* ng_e = (e < 8) ? ng + (size_t)e * HDIM : sng;
    const float* nu_e = (e < 8) ? nu + (size_t)e * HDIM : snu;
    const float* nd_e = (e < 8) ? nd + (size_t)e * IDIM : snd;
    int mig = (e < 8) ? e*3 + 0 : 24;
    int miu = (e < 8) ? e*3 + 1 : 25;
    int mid = (e < 8) ? e*3 + 2 : 26;
    const float* wrow = (e < 8) ? wtab + (size_t)e * T_TOK : nullptr;

    actquant_kernel<<<T_TOK, 256, 0, stream>>>(x, HDIM, HDIM/256, ng_e, nu_e,
                                               xqg, xqu, asg, asu);
    gemm1_kernel<<<dim3(T_TOK/128, IDIM/64), 256, 0, stream>>>(
        xqg, xqu, asg, asu, qw + (size_t)mig*WN, qw + (size_t)miu*WN,
        wscale, mig, miu, hidden);
    actquant_kernel<<<T_TOK, 256, 0, stream>>>(hidden, IDIM, IDIM/256, nd_e, nullptr,
                                               hq, nullptr, hs, nullptr);
    gemm2_kernel<<<dim3(T_TOK/128, HDIM/64), 256, 0, stream>>>(
        hq, hs, qw + (size_t)mid*WN, wscale, mid, wrow, out);
  }
  clip_kernel<<<(T_TOK*HDIM + 255)/256, 256, 0, stream>>>(out, T_TOK*HDIM);
}
